// MoEDenseActDense_35983236005998
// MI455X (gfx1250) — compile-verified, matrix operand order checked by reading, not run on hardware
//
#include <hip/hip_runtime.h>
#include <hip/hip_bf16.h>

// ---------------- problem constants ----------------
#define NTOK 8192   // B*S
#define DD   1024   // model dim
#define EE   8      // experts
#define HH   512    // hidden
#define OO   1024   // output dim
#define TOPK 4
#define MT   64     // tokens per FFN tile

typedef __attribute__((ext_vector_type(16))) __bf16 v16bf;
typedef __attribute__((ext_vector_type(8)))  __bf16 bf16x8;
typedef __attribute__((ext_vector_type(4)))  __bf16 bf16x4;
typedef __attribute__((ext_vector_type(8)))  float  v8f;

// ---------------- kernel 0: zero output + expert counters ----------------
__global__ void moe_zero_kernel(float* __restrict__ out, int n4, int* __restrict__ counts) {
    int i = blockIdx.x * blockDim.x + threadIdx.x;
    if (i < n4) ((float4*)out)[i] = make_float4(0.f, 0.f, 0.f, 0.f);
    if (blockIdx.x == 0 && threadIdx.x < EE) counts[threadIdx.x] = 0;
}

// ---------------- kernel 1: x (f32) -> bf16, row-major ----------------
__global__ void moe_cvtx_kernel(const float* __restrict__ x, __bf16* __restrict__ xb, int n4) {
    int i = blockIdx.x * blockDim.x + threadIdx.x;
    if (i >= n4) return;
    float4 v = ((const float4*)x)[i];
    bf16x4 o;
    o.x = (__bf16)v.x; o.y = (__bf16)v.y; o.z = (__bf16)v.z; o.w = (__bf16)v.w;
    ((bf16x4*)xb)[i] = o;
}

// ---------------- kernel 2: swizzle weights into WMMA B-fragment layout --------
// src: f32 [E][NN][KK] row-major (y = x @ W^T: N = output col, K = reduce dim)
// dst: bf16 fragments; frag f = ((e*(NN/16)+nt)*(KK/32)+kt); per frag 32 lanes x 16 bf16,
// each lane's 16 values contiguous (32B) -> two clean b128 loads in the GEMM.
// Lane l holds column n = nt*16 + (l&15); K pattern mirrors the 16-bit A layout:
// lanes 0-15: K = {0..7, 16..23}; lanes 16-31: K = {8..15, 24..31} (pairs per VGPR).
__global__ void moe_swzw_kernel(const float* __restrict__ src, __bf16* __restrict__ dst,
                                int NN, int KK) {
    int gid  = blockIdx.x * blockDim.x + threadIdx.x;
    int lane = gid & 31;
    int f    = gid >> 5;
    int fragsPerE = (NN / 16) * (KK / 32);
    if (f >= EE * fragsPerE) return;
    int e  = f / fragsPerE;
    int r  = f % fragsPerE;
    int nt = r / (KK / 32);
    int kt = r % (KK / 32);
    int n  = nt * 16 + (lane & 15);
    const float* srow = src + ((size_t)e * NN + n) * KK + kt * 32;
    int kadd = (lane >= 16) ? 8 : 0;
    v16bf vals;
#pragma unroll
    for (int v = 0; v < 8; ++v) {
#pragma unroll
        for (int p = 0; p < 2; ++p) {
            int kb = (v < 4) ? (2 * v + p) : (16 + 2 * (v - 4) + p);
            vals[2 * v + p] = (__bf16)srow[kb + kadd];
        }
    }
    *(v16bf*)(dst + (size_t)f * 512 + lane * 16) = vals;
}

// ---------------- kernel 3: gating + top-4 routing ----------------
__global__ void moe_gate_kernel(const float* __restrict__ x, const float* __restrict__ wg,
                                int* __restrict__ counts, int* __restrict__ tok) {
    int wave = threadIdx.x >> 5, lane = threadIdx.x & 31;
    int t = blockIdx.x * (blockDim.x >> 5) + wave;
    if (t >= NTOK) return;
    float acc[EE];
#pragma unroll
    for (int e = 0; e < EE; ++e) acc[e] = 0.f;
    const float* xr = x + (size_t)t * DD;
    for (int d = lane; d < DD; d += 32) {
        float xv = xr[d];
#pragma unroll
        for (int e = 0; e < EE; ++e) acc[e] += xv * wg[e * DD + d];
    }
#pragma unroll
    for (int e = 0; e < EE; ++e)
        for (int off = 16; off > 0; off >>= 1) acc[e] += __shfl_xor(acc[e], off, 32);
    if (lane == 0) {
        unsigned chosen = 0;
        for (int k = 0; k < TOPK; ++k) {
            int best = 0; float bv = -__builtin_inff();
#pragma unroll
            for (int e = 0; e < EE; ++e)
                if (!((chosen >> e) & 1u) && acc[e] > bv) { bv = acc[e]; best = e; }
            chosen |= (1u << best);
            int pos = atomicAdd(&counts[best], 1);
            tok[best * NTOK + pos] = t;
        }
    }
}

// ---------------- A-fragment loader from LDS (row-major bf16 tile) -------------
__device__ __forceinline__ v16bf load_a_lds(const __bf16* base, int pitch, int rb,
                                            int kt, int lane) {
    int lrow = lane & 15;
    const __bf16* p = base + (size_t)(rb * 16 + lrow) * pitch + kt * 32 + ((lane >= 16) ? 8 : 0);
    union { v16bf v; bf16x8 h[2]; } u;
    u.h[0] = *(const bf16x8*)p;
    u.h[1] = *(const bf16x8*)(p + 16);
    return u.v;
}

// ---------------- software-pipelined 16xK * Kx128 tile GEMM --------------------
// Double-buffered B fragments: loads for phase k+1 issue before the 8 WMMAs of
// phase k, so the backend can wait with slack (loadcnt<=16) and no WMMA WAR nops.
__device__ __forceinline__ void gemm8(const __bf16* __restrict__ Als, int pitch, int KT,
                                      const __bf16* __restrict__ wbase, int rb, int lane,
                                      v8f acc[8]) {
    const size_t jstride = (size_t)KT * 512;  // elements between adjacent n-tiles
    v16bf b0[8], b1[8];
#pragma unroll
    for (int j = 0; j < 8; ++j) b0[j] = *(const v16bf*)(wbase + (size_t)j * jstride);
#pragma unroll 1
    for (int kt = 0; kt < KT; kt += 2) {
        v16bf a0 = load_a_lds(Als, pitch, rb, kt, lane);
        const __bf16* p1 = wbase + (size_t)(kt + 1) * 512;
#pragma unroll
        for (int j = 0; j < 8; ++j) b1[j] = *(const v16bf*)(p1 + (size_t)j * jstride);
#pragma unroll
        for (int j = 0; j < 8; ++j)
            acc[j] = __builtin_amdgcn_wmma_f32_16x16x32_bf16(
                false, a0, false, b0[j], (short)0, acc[j], false, false);
        v16bf a1 = load_a_lds(Als, pitch, rb, kt + 1, lane);
        int kn = (kt + 2 < KT) ? (kt + 2) : 0;  // harmless redundant load on last phase
        const __bf16* p2 = wbase + (size_t)kn * 512;
#pragma unroll
        for (int j = 0; j < 8; ++j) b0[j] = *(const v16bf*)(p2 + (size_t)j * jstride);
#pragma unroll
        for (int j = 0; j < 8; ++j)
            acc[j] = __builtin_amdgcn_wmma_f32_16x16x32_bf16(
                false, a1, false, b1[j], (short)0, acc[j], false, false);
    }
}

// ---------------- kernel 4: fused per-expert FFN (bf16 WMMA) ----------------
__global__ void moe_ffn_kernel(const __bf16* __restrict__ xb,
                               const __bf16* __restrict__ w1s,
                               const __bf16* __restrict__ w2s,
                               const int* __restrict__ counts,
                               const int* __restrict__ tok,
                               float* __restrict__ out) {
    extern __shared__ char smem[];
    __bf16* Xs = (__bf16*)smem;                 // 64 x 1024 bf16 = 128 KB (LDS offset 0)
    __bf16* Hs = (__bf16*)(smem + MT * DD * 2); // 64 x 512  bf16 =  64 KB

    const int e = blockIdx.x;
    const int cnt = counts[e];
    const int start = blockIdx.y * MT;
    if (start >= cnt) return;

    const int tid = threadIdx.x;
    // gather 64 token rows of bf16 x straight into LDS via async copies
    {
        int row = tid >> 2, part = tid & 3;
        int gidx = start + row;
        int tk = (gidx < cnt) ? tok[e * NTOK + gidx] : tok[e * NTOK + start];
        unsigned loff = (unsigned)((row * DD + part * 256) * 2); // LDS byte offset (Xs @ 0)
        unsigned long long ga =
            (unsigned long long)(const void*)(xb + (size_t)tk * DD + part * 256);
#pragma unroll 8
        for (int i = 0; i < 32; ++i) {
            unsigned lo = loff + i * 16;
            unsigned long long g = ga + (unsigned long long)i * 16;
            asm volatile("global_load_async_to_lds_b128 %0, %1, off"
                         :: "v"(lo), "v"(g) : "memory");
        }
        asm volatile("s_wait_asynccnt 0" ::: "memory");
    }
    __syncthreads();

    const int wave = tid >> 5, lane = tid & 31;
    const int rb = wave >> 1, half = wave & 1;
    const int lrow = lane & 15;
    const int hiL = (lane >= 16);

    // ---------------- stage 1: h = relu(X @ W1^T) ----------------
    for (int pass = 0; pass < 2; ++pass) {
        v8f acc[8];
#pragma unroll
        for (int j = 0; j < 8; ++j) acc[j] = (v8f)0.f;
        const int nt0 = half * 16 + pass * 8; // of H/16 = 32 column tiles
        const __bf16* wbase =
            w1s + (((size_t)e * (HH / 16) + nt0) * (DD / 32)) * 512 + lane * 16;
        gemm8(Xs, DD, DD / 32, wbase, rb, lane, acc);
#pragma unroll
        for (int j = 0; j < 8; ++j) {
#pragma unroll
            for (int i = 0; i < 8; ++i) {
                float v = acc[j][i];
                v = v > 0.f ? v : 0.f;
                int M = rb * 16 + (hiL ? (i + 8) : i);
                int col = (nt0 + j) * 16 + lrow;
                Hs[(size_t)M * HH + col] = (__bf16)v;
            }
        }
    }
    __syncthreads();

    // ---------------- stage 2: y = h @ W2^T, scatter-add ----------------
    for (int pass = 0; pass < 4; ++pass) {
        v8f acc[8];
#pragma unroll
        for (int j = 0; j < 8; ++j) acc[j] = (v8f)0.f;
        const int nt0 = half * 32 + pass * 8; // of O/16 = 64 column tiles
        const __bf16* wbase =
            w2s + (((size_t)e * (OO / 16) + nt0) * (HH / 32)) * 512 + lane * 16;
        gemm8(Hs, HH, HH / 32, wbase, rb, lane, acc);
#pragma unroll
        for (int j = 0; j < 8; ++j) {
#pragma unroll
            for (int i = 0; i < 8; ++i) {
                int M = rb * 16 + (hiL ? (i + 8) : i);
                int g = start + M;
                if (g < cnt) {
                    int tk = tok[e * NTOK + g];
                    atomicAdd(out + (size_t)tk * OO + (nt0 + j) * 16 + lrow, acc[j][i]);
                }
            }
        }
    }
}

// ---------------- launcher ----------------
extern "C" void kernel_launch(void* const* d_in, const int* in_sizes, int n_in,
                              void* d_out, int out_size, void* d_ws, size_t ws_size,
                              hipStream_t stream) {
    (void)in_sizes; (void)n_in; (void)out_size; (void)ws_size;
    const float* x  = (const float*)d_in[0];
    const float* wg = (const float*)d_in[1];
    const float* w1 = (const float*)d_in[2];
    const float* w2 = (const float*)d_in[3];
    float* out = (float*)d_out;

    char* ws = (char*)d_ws;
    int*    counts = (int*)ws;
    int*    tok    = (int*)(ws + 1024);
    size_t  off_xb = (size_t)1 << 20;                        // 1 MB
    size_t  off_w1 = off_xb + (size_t)NTOK * DD * 2;         // +16 MB
    size_t  off_w2 = off_w1 + (size_t)EE * HH * DD * 2;      // +8 MB
    __bf16* xb  = (__bf16*)(ws + off_xb);
    __bf16* w1s = (__bf16*)(ws + off_w1);
    __bf16* w2s = (__bf16*)(ws + off_w2);

    moe_zero_kernel<<<(NTOK * OO / 4 + 255) / 256, 256, 0, stream>>>(out, NTOK * OO / 4, counts);
    moe_cvtx_kernel<<<(NTOK * DD / 4 + 255) / 256, 256, 0, stream>>>(x, xb, NTOK * DD / 4);
    moe_swzw_kernel<<<1024, 256, 0, stream>>>(w1, w1s, HH, DD);
    moe_swzw_kernel<<<1024, 256, 0, stream>>>(w2, w2s, OO, HH);
    moe_gate_kernel<<<NTOK / 8, 256, 0, stream>>>(x, wg, counts, tok);
    moe_ffn_kernel<<<dim3(EE, NTOK / MT), 256, MT * DD * 2 + MT * HH * 2, stream>>>(
        xb, w1s, w2s, counts, tok, out);
}
